// FastHebbMemory_48017734369417
// MI455X (gfx1250) — matrix operand order repaired and checked by reading, not compiled, not run on hardware
//
#include <hip/hip_runtime.h>

// ---------------------------------------------------------------------------
// FastHebbMemory retrieval for MI455X (gfx1250, wave32, WMMA).
//   logits = normalize(q) @ K^T / tau - beta*(T-t)^2/(2 sigma^2)
//   out    = softmax(top32(logits)) @ V[top32 idx]
//
// v4: 128x128 block tiles, persistent accumulators, double-buffered
// GLOBAL_LOAD_ASYNC_TO_LDS slab staging (ASYNCcnt), and explicit
// sched_group_barrier interleaving (DS-read / WMMA groups) so B-fragment
// LDS loads run 2 tiles ahead of the WMMA pipe instead of the serialized
// load -> s_wait_dscnt 0 -> wmma chain the default scheduler emits.
// ---------------------------------------------------------------------------

typedef __attribute__((ext_vector_type(16))) __bf16 v16bf;
typedef __attribute__((ext_vector_type(8)))  __bf16 v8bf;
typedef __attribute__((ext_vector_type(4)))  __bf16 v4bf;
typedef __attribute__((ext_vector_type(8)))  float  v8f;

#define TAU_INV   5.0f     // 1/0.2
#define BETA      0.5f
#define SIGMA     1.0f
#define QTIME     100.0f
#define KTOP      32
#define NSPLIT    32       // N-dimension split for the top-k pass
#define TQ        128      // queries per block
#define TK        128      // keys per block-iteration
#define DS        64       // D-dimension slab depth (double-buffered)

__device__ __forceinline__ float neg_inf() { return -__builtin_inff(); }

#define WAIT_ASYNC() asm volatile("s_wait_asynccnt 0x0" ::: "memory")
// scheduling groups: 0x100 = DS read, 0x008 = MFMA/WMMA
#define SG(mask, size) __builtin_amdgcn_sched_group_barrier((mask), (size), 0)

// one 16-byte async global->LDS transfer (ASYNCcnt-tracked, no VGPR data)
__device__ __forceinline__ void async_copy16(unsigned lds_off, const void* gaddr) {
    asm volatile("global_load_async_to_lds_b128 %0, %1, off"
                 :: "v"(lds_off), "v"((unsigned long long)(uintptr_t)gaddr)
                 : "memory");
}

// ---------------- Phase 1a: normalize query rows -> bf16 -------------------
__global__ void qnorm_bf16_kernel(const float* __restrict__ q,
                                  __bf16* __restrict__ qbf, int D) {
    const int b = blockIdx.x, tid = threadIdx.x;
    __shared__ float red[256];
    const float* row = q + (size_t)b * D;
    float ss = 0.f;
    for (int d = tid; d < D; d += 256) { float v = row[d]; ss += v * v; }
    red[tid] = ss;
    __syncthreads();
    for (int s = 128; s > 0; s >>= 1) {
        if (tid < s) red[tid] += red[tid + s];
        __syncthreads();
    }
    const float scale = 1.0f / fmaxf(sqrtf(red[0]), 1e-12f);
    __bf16* orow = qbf + (size_t)b * D;
    for (int d = tid; d < D; d += 256) orow[d] = (__bf16)(row[d] * scale);
}

// ---------------- Phase 1b: K f32 -> bf16 (L2-resident bank) ---------------
__global__ void kcvt_bf16_kernel(const float* __restrict__ K,
                                 __bf16* __restrict__ kbf, size_t n4) {
    const size_t i = (size_t)blockIdx.x * blockDim.x + threadIdx.x;
    if (i < n4) {
        const float4 v = ((const float4*)K)[i];
        v4bf o = { (__bf16)v.x, (__bf16)v.y, (__bf16)v.z, (__bf16)v.w };
        ((v4bf*)kbf)[i] = o;
    }
}

// ---------------- Phase 1c: recency penalty per key ------------------------
__global__ void penalty_kernel(const float* __restrict__ times,
                               float* __restrict__ pen, int N) {
    const int i = blockIdx.x * blockDim.x + threadIdx.x;
    if (i < N) {
        const float dt = QTIME - times[i];
        pen[i] = (BETA / (2.0f * SIGMA * SIGMA)) * dt * dt;
    }
}

// ---------------- Phase 2: WMMA GEMM + streaming top-32 --------------------
// grid: (B/TQ, NSPLIT); block: 256 threads (8 waves).
// Per key-iteration the block computes a TQ x TK logit tile: wave w owns
// query rows [16w, 16w+16) and all TK key columns (8 accumulator tiles).
// D is consumed in DS-deep slabs, async-staged into ping-pong LDS buffers.
__global__ void __launch_bounds__(256)
gemm_topk_kernel(const __bf16* __restrict__ qbf,
                 const __bf16* __restrict__ kbf,
                 const float*  __restrict__ pen,
                 float* __restrict__ candv,
                 int*   __restrict__ candi,
                 int D, int chunk, int nsplit) {
    // 64KB: two ping-pong {Q,K} slab buffers, reused as the score tile
    // layout: [Q0 16K][K0 16K][Q1 16K][K1 16K]
    __shared__ __align__(16) char lds_raw[4 * TQ * DS * 2];
    float* sS = (float*)lds_raw;                         // [TQ][TK] 64KB
    __shared__ float sPen[TK];
    __shared__ float topv[TQ][KTOP];                     // 16KB
    __shared__ int   topi[TQ][KTOP];                     // 16KB

    const int tid  = threadIdx.x;
    const int wave = tid >> 5;
    const int lane = tid & 31;
    const int hi   = lane >> 4;     // half-wave select (K-range split)
    const int m16  = lane & 15;     // A-row / B-column within a 16x16 tile
    const int q0   = blockIdx.x * TQ;
    const int n0   = blockIdx.y * chunk;

    const unsigned lds_base = (unsigned)(uintptr_t)lds_raw; // LDS byte offset

    if (tid < TQ) {
        #pragma unroll
        for (int s = 0; s < KTOP; s++) { topv[tid][s] = neg_inf(); topi[tid][s] = 0; }
    }
    float curmin = neg_inf();
    int   minslot = 0;

    const int nslab = D / DS;
    // per-thread staging coordinates: 1024 16B vectors per operand per slab
    const int srow = tid >> 3;           // +32 rows per staging step
    const int scol = (tid & 7) * 8;      // DS/8 = 8 vectors per row

    for (int it = 0; it < chunk; it += TK) {
        const int k0 = n0 + it;
        if (tid < TK) sPen[tid] = pen[k0 + tid];

        v8f acc[8];
        #pragma unroll
        for (int t = 0; t < 8; t++) acc[t] = {};

        const __bf16* gq = qbf + (size_t)q0 * D;
        const __bf16* gk = kbf + (size_t)k0 * D;

        // prologue: async-stage slab 0 into buffer 0
        #pragma unroll
        for (int r = 0; r < 4; r++) {
            const int row = srow + r * 32;
            async_copy16(lds_base + (row * DS + scol) * 2,
                         gq + (size_t)row * D + scol);
            async_copy16(lds_base + 16384 + (row * DS + scol) * 2,
                         gk + (size_t)row * D + scol);
        }

        for (int sl = 0; sl < nslab; sl++) {
            const int cur = sl & 1;
            WAIT_ASYNC();          // my slab-sl writes to LDS are done
            __syncthreads();       // everyone's are; prior readers finished

            if (sl + 1 < nslab) {  // async-stage next slab into the other buf
                const int d0 = (sl + 1) * DS;
                const unsigned nb = lds_base + ((sl + 1) & 1) * 32768;
                #pragma unroll
                for (int r = 0; r < 4; r++) {
                    const int row = srow + r * 32;
                    async_copy16(nb + (row * DS + scol) * 2,
                                 gq + (size_t)row * D + d0 + scol);
                    async_copy16(nb + 16384 + (row * DS + scol) * 2,
                                 gk + (size_t)row * D + d0 + scol);
                }
            }

            const __bf16* aQ = (const __bf16*)(lds_raw + cur * 32768);
            const __bf16* aK = (const __bf16*)(lds_raw + cur * 32768 + 16384);
            // A-fragment: lane<16 -> K 0..7 & 16..23, lane>=16 -> 8..15 & 24..31
            const __bf16* arow = aQ + (wave * 16 + m16) * DS + hi * 8;
            // B-fragment: lane<16 -> K 0..7 & 16..23, lane>=16 -> 8..15 & 24..31
            const __bf16* brow = aK + m16 * DS + hi * 8;

            #pragma unroll
            for (int s = 0; s < DS; s += 32) {
                v8bf alo = *(const v8bf*)(arow + s);
                v8bf ahi = *(const v8bf*)(arow + s + 16);
                v16bf a = __builtin_shufflevector(alo, ahi,
                    0,1,2,3,4,5,6,7,8,9,10,11,12,13,14,15);
                v16bf b[8];
                #pragma unroll
                for (int t = 0; t < 8; t++) {   // batch B loads first
                    v8bf blo = *(const v8bf*)(brow + t * 16 * DS + s);
                    v8bf bhi = *(const v8bf*)(brow + t * 16 * DS + s + 16);
                    b[t] = __builtin_shufflevector(blo, bhi,
                        0,1,2,3,4,5,6,7,8,9,10,11,12,13,14,15);
                }
                #pragma unroll
                for (int t = 0; t < 8; t++)     // then 8 independent WMMAs
                    acc[t] = __builtin_amdgcn_wmma_f32_16x16x32_bf16(
                                 false, a, false, b[t], (short)0, acc[t],
                                 false, false);

                // force a pipelined issue order per 32-deep k-step:
                //   [A + B0 + B1 loads][WMMA, 2 DS reads]x6[2 WMMAs]
                SG(0x100, 6);               // 6 DS reads (A pair, B0, B1)
                #pragma unroll
                for (int g = 0; g < 6; g++) {
                    SG(0x008, 1);           // 1 WMMA
                    SG(0x100, 2);           // next B-fragment pair
                }
                SG(0x008, 2);               // trailing 2 WMMAs
            }
        }
        __syncthreads();   // all waves finished reading the last slab

        // spill logits to the (reused) score tile with scale + per-key penalty
        #pragma unroll
        for (int t = 0; t < 8; t++) {
            const float p = sPen[t * 16 + m16];
            #pragma unroll
            for (int i = 0; i < 8; i++)
                sS[(wave * 16 + i + 8 * hi) * TK + t * 16 + m16] =
                    acc[t][i] * TAU_INV - p;
        }
        __syncthreads();

        // streaming top-32: one owner thread per query row
        if (tid < TQ) {
            for (int j = 0; j < TK; j++) {
                const float v = sS[tid * TK + j];
                if (v > curmin) {
                    topv[tid][minslot] = v;
                    topi[tid][minslot] = k0 + j;
                    curmin = topv[tid][0]; minslot = 0;
                    #pragma unroll
                    for (int s2 = 1; s2 < KTOP; s2++) {
                        const float tv = topv[tid][s2];
                        if (tv < curmin) { curmin = tv; minslot = s2; }
                    }
                }
            }
        }
        __syncthreads();   // scan done before next iteration restages/writes
    }

    if (tid < TQ) {
        const size_t base = ((size_t)(q0 + tid) * nsplit + blockIdx.y) * KTOP;
        #pragma unroll
        for (int s = 0; s < KTOP; s++) {
            candv[base + s] = topv[tid][s];
            candi[base + s] = topi[tid][s];
        }
    }
}

// ---------------- Phase 3: merge candidates + softmax + V gather -----------
__global__ void merge_out_kernel(const float* __restrict__ candv,
                                 const int*   __restrict__ candi,
                                 const float* __restrict__ V,
                                 float* __restrict__ out,
                                 int D, int M) {
    __shared__ float cv[NSPLIT * KTOP];
    __shared__ int   ci[NSPLIT * KTOP];
    __shared__ float rv[256];
    __shared__ int   rp[256];
    __shared__ float selv[KTOP];
    __shared__ int   seli[KTOP];
    __shared__ float attn[KTOP];

    const int b = blockIdx.x, tid = threadIdx.x;
    for (int j = tid; j < M; j += 256) {
        cv[j] = candv[(size_t)b * M + j];
        ci[j] = candi[(size_t)b * M + j];
    }
    __syncthreads();

    for (int k = 0; k < KTOP; k++) {
        float bv = neg_inf(); int bp = 0;
        for (int j = tid; j < M; j += 256)
            if (cv[j] > bv) { bv = cv[j]; bp = j; }
        rv[tid] = bv; rp[tid] = bp;
        __syncthreads();
        for (int s = 128; s > 0; s >>= 1) {
            if (tid < s && rv[tid + s] > rv[tid]) { rv[tid] = rv[tid + s]; rp[tid] = rp[tid + s]; }
            __syncthreads();
        }
        if (tid == 0) {
            selv[k] = rv[0];
            seli[k] = ci[rp[0]];
            cv[rp[0]] = neg_inf();
        }
        __syncthreads();
    }

    if (tid < KTOP) attn[tid] = __expf(selv[tid] - selv[0]);
    __syncthreads();
    float ssum = 0.f;
    #pragma unroll
    for (int k = 0; k < KTOP; k++) ssum += attn[k];
    const float inv = 1.0f / ssum;

    for (int d0 = tid * 4; d0 < D; d0 += 256 * 4) {
        float4 acc = make_float4(0.f, 0.f, 0.f, 0.f);
        #pragma unroll
        for (int k = 0; k < KTOP; k++) {
            const float w = attn[k] * inv;
            const float4 vv = *(const float4*)(V + (size_t)seli[k] * D + d0);
            acc.x += w * vv.x; acc.y += w * vv.y;
            acc.z += w * vv.z; acc.w += w * vv.w;
        }
        *(float4*)(out + (size_t)b * D + d0) = acc;
    }
}

// ---------------------------------------------------------------------------
extern "C" void kernel_launch(void* const* d_in, const int* in_sizes, int n_in,
                              void* d_out, int out_size, void* d_ws, size_t ws_size,
                              hipStream_t stream) {
    const float* q     = (const float*)d_in[0];
    const float* K     = (const float*)d_in[1];
    const float* V     = (const float*)d_in[2];
    const float* times = (const float*)d_in[3];
    // topk (d_in[4]) is a device scalar == 32; KTOP hardcoded to match.

    const int N = in_sizes[3];
    const int D = in_sizes[1] / N;   // 1024
    const int B = in_sizes[0] / D;   // 1024

    // workspace layout:
    //   Kbf   : N*D bf16   (128 MB)  -- L2-resident similarity bank
    //   Qbf   : B*D bf16   (  2 MB)
    //   pen   : N   f32    (256 KB)
    //   candv : B*NSPLIT*32 f32 (4 MB)
    //   candi : B*NSPLIT*32 i32 (4 MB)
    char* w = (char*)d_ws;
    __bf16* Kbf = (__bf16*)w;                 w += (size_t)N * D * sizeof(__bf16);
    __bf16* Qbf = (__bf16*)w;                 w += (size_t)B * D * sizeof(__bf16);
    float*  pen = (float*)w;                  w += (size_t)N * sizeof(float);
    float*  candv = (float*)w;                w += (size_t)B * NSPLIT * KTOP * sizeof(float);
    int*    candi = (int*)w;

    // Phase 1: conversions / precompute
    qnorm_bf16_kernel<<<B, 256, 0, stream>>>(q, Qbf, D);
    {
        const size_t n4 = (size_t)N * D / 4;
        const int blocks = (int)((n4 + 255) / 256);
        kcvt_bf16_kernel<<<blocks, 256, 0, stream>>>(K, Kbf, n4);
    }
    penalty_kernel<<<(N + 255) / 256, 256, 0, stream>>>(times, pen, N);

    // Phase 2: fused WMMA GEMM + streaming top-32
    const int chunk = N / NSPLIT;               // 2048 keys per block
    dim3 grid2(B / TQ, NSPLIT);                 // (8, 32)
    gemm_topk_kernel<<<grid2, 256, 0, stream>>>(Qbf, Kbf, pen, candv, candi,
                                                D, chunk, NSPLIT);

    // Phase 3: merge + softmax + weighted V gather
    merge_out_kernel<<<B, 256, 0, stream>>>(candv, candi, V, (float*)d_out,
                                            D, NSPLIT * KTOP);
}
